// GraphMultiHeadedAttention_7224134992243
// MI455X (gfx1250) — matrix operand order repaired and checked by reading, not compiled, not run on hardware
//
#include <hip/hip_runtime.h>
#include <hip/hip_bf16.h>
#include <cfloat>

// GraphMultiHeadedAttention for MI455X (gfx1250), bf16 WMMA pipeline.
// N=8192 tokens, D=1024 features, fp32 in/out, bf16 tensor-core math.
// Single-pass flash-attention: no N x N materialization, no stats prepass.

#define NTOK 8192
#define DDIM 1024

typedef __bf16 v16bf __attribute__((ext_vector_type(16)));
typedef float  v8f   __attribute__((ext_vector_type(8)));

union Frag {
    v16bf v;
    unsigned short s[16];
    uint4 q[2];
};

__device__ __forceinline__ unsigned short f2bf(float f) {
    unsigned int u = __float_as_uint(f);
    unsigned int r = (u + 0x7FFFu + ((u >> 16) & 1u)) >> 16;
    return (unsigned short)r;
}

// pack 8 f32 (two float4) -> 8 bf16 in a uint4 (round-to-nearest-even)
__device__ __forceinline__ uint4 cvt8(const float4 a, const float4 b) {
    uint4 r;
    r.x = (unsigned)f2bf(a.x) | ((unsigned)f2bf(a.y) << 16);
    r.y = (unsigned)f2bf(a.z) | ((unsigned)f2bf(a.w) << 16);
    r.z = (unsigned)f2bf(b.x) | ((unsigned)f2bf(b.y) << 16);
    r.w = (unsigned)f2bf(b.z) | ((unsigned)f2bf(b.w) << 16);
    return r;
}

__device__ __forceinline__ v8f wmma_bf16(const Frag& a, const Frag& b, v8f c) {
    return __builtin_amdgcn_wmma_f32_16x16x32_bf16(
        /*neg_a=*/false, a.v, /*neg_b=*/false, b.v,
        /*c_mod=*/(short)0, c, /*reuse_a=*/false, /*reuse_b=*/false);
}

// Async global -> LDS copy of 16 bytes (GLOBAL_LOAD_ASYNC_TO_LDS_B128,
// tracked with ASYNCcnt). lds_dst low 32 bits == LDS byte offset.
__device__ __forceinline__ void async_ld16(void* lds_dst, const void* gsrc) {
    unsigned off = (unsigned)(size_t)lds_dst;
    asm volatile("global_load_async_to_lds_b128 %0, %1, off"
                 :: "v"(off), "v"(gsrc) : "memory");
}
__device__ __forceinline__ void wait_async0() {
    asm volatile("s_wait_asynccnt 0" ::: "memory");
}

// ---------------------------------------------------------------------------
// Kernel 0: elementwise f32 -> bf16 convert (inputs & weights), 8 elem/thread
// ---------------------------------------------------------------------------
__global__ __launch_bounds__(256)
void cvt_bf16_kernel(const float* __restrict__ src, unsigned short* __restrict__ dst,
                     int n8)
{
    const int i = blockIdx.x * 256 + threadIdx.x;
    if (i < n8) {
        const float4* s = (const float4*)(src + (size_t)i * 8);
        ((uint4*)dst)[i] = cvt8(s[0], s[1]);
    }
}

// ---------------------------------------------------------------------------
// Kernel 1/3: Out = (A @ W^T + bias) * scale     (A, W already bf16)
//   A: [N, D] bf16, W: [D, D] bf16 (row-major, row = output feature),
//   bias: [D] f32.  Out: bf16 [N,D], bf16 transposed [D,N], or f32 [N,D].
// Block: 128 threads (4 waves); wave computes a 16x64 strip (4 WMMA tiles
// sharing one A fragment -> half the load traffic). Grid: (D/256, N/16).
// ---------------------------------------------------------------------------
template <bool OUT_F32, bool TRANSPOSE_OUT>
__global__ __launch_bounds__(128)
void proj_kernel(const unsigned short* __restrict__ A,
                 const unsigned short* __restrict__ W,
                 const float* __restrict__ bias, void* __restrict__ Out,
                 float scale)
{
    const int lane = threadIdx.x & 31;
    const int wave = threadIdx.x >> 5;
    const int lr   = lane & 15;
    const int half = lane >> 4;
    const int col0 = (blockIdx.x * 4 + wave) * 64;   // 64-wide output strip
    const int row0 = blockIdx.y * 16;

    const unsigned short* Ap0 = A + (size_t)(row0 + lr) * DDIM;

    v8f acc[4];
    {
        v8f z = {};
#pragma unroll
        for (int c = 0; c < 4; c++) acc[c] = z;
    }

    for (int k0 = 0; k0 < DDIM; k0 += 32) {
        Frag a;
        // A operand, contiguous k order (same as B): lanes 0-15 K 0..15, lanes 16-31 K 16..31
        const uint4* Ap = (const uint4*)(Ap0 + k0 + half * 16);
        a.q[0] = Ap[0];
        a.q[1] = Ap[1];
#pragma unroll
        for (int c = 0; c < 4; c++) {
            Frag b;
            // B layout: lanes 0-15 K 0..15, lanes 16-31 K 16..31
            const uint4* Wp = (const uint4*)(W + (size_t)(col0 + c * 16 + lr) * DDIM
                                             + k0 + half * 16);
            b.q[0] = Wp[0];
            b.q[1] = Wp[1];
            acc[c] = wmma_bf16(a, b, acc[c]);
        }
    }

#pragma unroll
    for (int c = 0; c < 4; c++) {
        const int col = col0 + c * 16 + lr;          // N = lane % 16
        const float bv = bias[col];
#pragma unroll
        for (int r = 0; r < 8; r++) {
            const int row = row0 + r + half * 8;     // C layout: VGPR r -> M
            const float y = (acc[c][r] + bv) * scale;
            if (OUT_F32) {
                ((float*)Out)[(size_t)row * DDIM + col] = y;
            } else if (TRANSPOSE_OUT) {
                ((unsigned short*)Out)[(size_t)col * NTOK + row] = f2bf(y);
            } else {
                ((unsigned short*)Out)[(size_t)row * DDIM + col] = f2bf(y);
            }
        }
    }
}

// ---------------------------------------------------------------------------
// Kernel 2: fused flash attention.
//   X = softmax(mask(q k^T)) @ v, online softmax, output bf16 [N, D].
//   Block: 256 threads (8 waves) per 16-row query block.
//   Per 32-key block: waves split the D=1024 score reduction (128 each),
//   partials reduced via LDS; wave0 runs the online masked softmax
//   (running row max m and sum l), emits unnormalized bf16 P tile plus a
//   per-row rescale factor; every wave rescales its 16x128 output
//   accumulator and adds P @ V (V pre-transposed -> contiguous B loads).
//   Final 1/l normalization at the end.
// ---------------------------------------------------------------------------
__global__ __launch_bounds__(256)
void fused_attn_kernel(const unsigned short* __restrict__ qb,
                       const unsigned short* __restrict__ kb,
                       const unsigned short* __restrict__ vbT,
                       const float* __restrict__ adj,
                       unsigned short* __restrict__ xb)
{
    __shared__ unsigned short qs[16][1032];   // 33 KB Q tile (pitch-padded)
    __shared__ float sred[2][8][256];         // 16 KB partial score tiles
    __shared__ unsigned short ps[16][40];     // P tile 16x32 bf16, pitch 80B
    __shared__ float fac[16];                 // per-row rescale factor
    __shared__ float linv[16];                // per-row 1/l

    const int lane  = threadIdx.x & 31;
    const int wave  = threadIdx.x >> 5;
    const int lr    = lane & 15;
    const int half  = lane >> 4;
    const int q0    = blockIdx.x * 16;
    const int kbase = wave * 128;  // this wave's K-slice for scores
    const int d0    = wave * 128;  // this wave's feature chunk for output

    // stage Q tile (16 x 1024 bf16 = 32 KB) via async DMA path
    for (int i = threadIdx.x; i < 16 * 128; i += 256) {
        const int r = i >> 7, c = i & 127;
        async_ld16(&qs[r][c * 8], qb + (size_t)(q0 + r) * DDIM + c * 8);
    }
    wait_async0();
    __syncthreads();

    v8f oacc[8];
    {
        v8f z = {};
#pragma unroll
        for (int t = 0; t < 8; t++) oacc[t] = z;
    }
    float m[8], l[8];   // online softmax state (live in wave 0 only)
#pragma unroll
    for (int r = 0; r < 8; r++) { m[r] = -FLT_MAX; l[r] = 0.0f; }

    for (int j2 = 0; j2 < NTOK / 32; j2++) {
        // ---- phase A: partial scores for key tiles 2*j2, 2*j2+1 over K-slice
#pragma unroll
        for (int t = 0; t < 2; t++) {
            const int j = j2 * 2 + t;
            const unsigned short* kB = kb + (size_t)(j * 16 + lr) * DDIM + kbase;
            v8f acc = {};
#pragma unroll
            for (int s = 0; s < 4; s++) {
                Frag a, b;
                const int k0 = kbase + s * 32;
                a.q[0] = *(const uint4*)&qs[lr][k0 + half * 16];
                a.q[1] = *(const uint4*)&qs[lr][k0 + half * 16 + 8];
                const uint4* kp = (const uint4*)(kB + s * 32 + half * 16);
                b.q[0] = kp[0];
                b.q[1] = kp[1];
                acc = wmma_bf16(a, b, acc);
            }
#pragma unroll
            for (int r = 0; r < 8; r++) sred[t][wave][lane * 8 + r] = acc[r];
        }
        __syncthreads();

        // ---- wave0: online masked softmax over the 32 new columns
        if (wave == 0) {
            float sc[2][8];
#pragma unroll
            for (int t = 0; t < 2; t++) {
                const int col = (j2 * 2 + t) * 16 + lr;
                float sacc[8];
#pragma unroll
                for (int r = 0; r < 8; r++) sacc[r] = 0.0f;
#pragma unroll
                for (int w = 0; w < 8; w++) {
                    const float4 p0 = *(const float4*)&sred[t][w][lane * 8];
                    const float4 p1 = *(const float4*)&sred[t][w][lane * 8 + 4];
                    sacc[0] += p0.x; sacc[1] += p0.y; sacc[2] += p0.z; sacc[3] += p0.w;
                    sacc[4] += p1.x; sacc[5] += p1.y; sacc[6] += p1.z; sacc[7] += p1.w;
                }
#pragma unroll
                for (int r = 0; r < 8; r++) {
                    const int row = q0 + r + half * 8;
                    const float av = adj[(size_t)row * NTOK + col];
                    const float v = sacc[r] * av;
                    sc[t][r] = (fabsf(v) < 1e-5f) ? -1e9f : v;
                }
            }
#pragma unroll
            for (int r = 0; r < 8; r++) {
                // block max over 32 columns (both tiles, 16 lanes each)
                float t0 = fmaxf(sc[0][r], sc[1][r]);
                t0 = fmaxf(t0, __shfl_xor(t0, 1));
                t0 = fmaxf(t0, __shfl_xor(t0, 2));
                t0 = fmaxf(t0, __shfl_xor(t0, 4));
                t0 = fmaxf(t0, __shfl_xor(t0, 8));
                const float mn = fmaxf(m[r], t0);
                const float f  = __expf(m[r] - mn);     // rescale for old acc
                const float p0 = __expf(sc[0][r] - mn);
                const float p1 = __expf(sc[1][r] - mn);
                float psum = p0 + p1;
                psum += __shfl_xor(psum, 1);
                psum += __shfl_xor(psum, 2);
                psum += __shfl_xor(psum, 4);
                psum += __shfl_xor(psum, 8);
                l[r] = l[r] * f + psum;
                m[r] = mn;
                ps[r + half * 8][lr]      = f2bf(p0);
                ps[r + half * 8][16 + lr] = f2bf(p1);
                if (lr == 0) fac[r + half * 8] = f;
            }
        }
        __syncthreads();

        // ---- phase B: rescale old accumulator, then O += P(16x32) @ V
        {
#pragma unroll
            for (int r = 0; r < 8; r++) {
                const float f = fac[r + half * 8];
#pragma unroll
                for (int t = 0; t < 8; t++) oacc[t][r] *= f;
            }
            Frag a;
            a.q[0] = *(const uint4*)&ps[lr][half * 16];
            a.q[1] = *(const uint4*)&ps[lr][half * 16 + 8];
            const int krow = j2 * 32 + half * 16;
            __builtin_prefetch(vbT + (size_t)(d0 + lr) * NTOK + krow + 32, 0, 1);
#pragma unroll
            for (int t = 0; t < 8; t++) {
                Frag b;
                const uint4* vp = (const uint4*)(vbT + (size_t)(d0 + t * 16 + lr) * NTOK + krow);
                b.q[0] = vp[0];
                b.q[1] = vp[1];
                oacc[t] = wmma_bf16(a, b, oacc[t]);
            }
        }
    }

    // final normalization by 1/l
    if (wave == 0 && lr == 0) {
#pragma unroll
        for (int r = 0; r < 8; r++) linv[r + half * 8] = 1.0f / l[r];
    }
    __syncthreads();

#pragma unroll
    for (int r = 0; r < 8; r++) {
        const float nf = linv[r + half * 8];
        const int row = q0 + r + half * 8;
#pragma unroll
        for (int t = 0; t < 8; t++) {
            const int col = d0 + t * 16 + lr;
            xb[(size_t)row * DDIM + col] = f2bf(oacc[t][r] * nf);
        }
    }
}

// ---------------------------------------------------------------------------
extern "C" void kernel_launch(void* const* d_in, const int* in_sizes, int n_in,
                              void* d_out, int out_size, void* d_ws, size_t ws_size,
                              hipStream_t stream)
{
    const float* query = (const float*)d_in[0];
    const float* key_  = (const float*)d_in[1];
    const float* value = (const float*)d_in[2];
    const float* adj   = (const float*)d_in[3];
    const float* Wq    = (const float*)d_in[4];
    const float* bq    = (const float*)d_in[5];
    const float* Wk    = (const float*)d_in[6];
    const float* bk    = (const float*)d_in[7];
    const float* Wv    = (const float*)d_in[8];
    const float* bv    = (const float*)d_in[9];
    const float* Wm    = (const float*)d_in[10];
    const float* bm    = (const float*)d_in[11];

    // workspace layout (~120 MB)
    char* ws = (char*)d_ws;
    size_t off = 0;
    const size_t ND = (size_t)NTOK * DDIM;   // 8M elems
    const size_t DD = (size_t)DDIM * DDIM;   // 1M elems
    unsigned short* qb  = (unsigned short*)(ws + off); off += ND * 2;
    unsigned short* kb  = (unsigned short*)(ws + off); off += ND * 2;
    unsigned short* vbT = (unsigned short*)(ws + off); off += ND * 2;
    unsigned short* xb  = (unsigned short*)(ws + off); off += ND * 2;
    unsigned short* Xq  = (unsigned short*)(ws + off); off += ND * 2;
    unsigned short* Xk  = (unsigned short*)(ws + off); off += ND * 2;
    unsigned short* Xv  = (unsigned short*)(ws + off); off += ND * 2;
    unsigned short* Wqb = (unsigned short*)(ws + off); off += DD * 2;
    unsigned short* Wkb = (unsigned short*)(ws + off); off += DD * 2;
    unsigned short* Wvb = (unsigned short*)(ws + off); off += DD * 2;
    unsigned short* Wmb = (unsigned short*)(ws + off); off += DD * 2;

    // --- pre-convert inputs & weights to bf16 (one pass each) ---
    const int nd8 = (int)(ND / 8), dd8 = (int)(DD / 8);
    cvt_bf16_kernel<<<(nd8 + 255) / 256, 256, 0, stream>>>(query, Xq, nd8);
    cvt_bf16_kernel<<<(nd8 + 255) / 256, 256, 0, stream>>>(key_,  Xk, nd8);
    cvt_bf16_kernel<<<(nd8 + 255) / 256, 256, 0, stream>>>(value, Xv, nd8);
    cvt_bf16_kernel<<<(dd8 + 255) / 256, 256, 0, stream>>>(Wq, Wqb, dd8);
    cvt_bf16_kernel<<<(dd8 + 255) / 256, 256, 0, stream>>>(Wk, Wkb, dd8);
    cvt_bf16_kernel<<<(dd8 + 255) / 256, 256, 0, stream>>>(Wv, Wvb, dd8);
    cvt_bf16_kernel<<<(dd8 + 255) / 256, 256, 0, stream>>>(Wm, Wmb, dd8);

    const float qscale = 0.011048543456039806f;  // 1/sqrt(N), N = 8192

    const dim3 gp(DDIM / 256, NTOK / 16);
    // q = (query @ Wq^T + bq) / sqrt(N)  -> bf16 [N, D]
    proj_kernel<false, false><<<gp, 128, 0, stream>>>(Xq, Wqb, bq, qb, qscale);
    // k = key @ Wk^T + bk               -> bf16 [N, D]
    proj_kernel<false, false><<<gp, 128, 0, stream>>>(Xk, Wkb, bk, kb, 1.0f);
    // v = value @ Wv^T + bv             -> bf16 transposed [D, N]
    proj_kernel<false, true ><<<gp, 128, 0, stream>>>(Xv, Wvb, bv, vbT, 1.0f);

    // x = softmax(mask(q k^T)) @ v      -> bf16 [N, D]  (single fused pass)
    fused_attn_kernel<<<NTOK / 16, 256, 0, stream>>>(qb, kb, vbT, adj, xb);

    // out = x @ Wm^T + bm               -> f32 [N, D]
    proj_kernel<true, false><<<gp, 128, 0, stream>>>(xb, Wmb, bm, (float*)d_out, 1.0f);
}
